// VideoEncoder_14800457302117
// MI455X (gfx1250) — matrix operand order repaired and checked by reading, not compiled, not run on hardware
//
#include <hip/hip_runtime.h>
#include <hip/hip_bf16.h>
#include <math.h>

typedef __attribute__((ext_vector_type(16))) _Float16 v16h;
typedef __attribute__((ext_vector_type(8)))  _Float16 v8h;
typedef __attribute__((ext_vector_type(8)))  float    v8f;

#define EDIM  256
#define NTOK  3136          // tokens per batch item (16*14*14)
#define BATCH 4
#define MTOT  (BATCH*NTOK)  // 12544
#define KCONV 2304          // 3*3*16*16
#define NHEAD 4
#define DHEAD 64

// ---------------------------------------------------------------- f32 -> f16
__global__ void cvt_f16_kernel(const float* __restrict__ src,
                               _Float16* __restrict__ dst, int n) {
  int i = blockIdx.x * blockDim.x + threadIdx.x;
  int stride = gridDim.x * blockDim.x;
  for (; i < n; i += stride) dst[i] = (_Float16)src[i];
}

// --------------------------------------------------- conv3d as im2col WMMA
// A = conv weights (256 x 2304) f16, B = im2col patch (on the fly from video)
// One block = 16 tokens; each of 4 waves covers 64 output channels.
__global__ void conv_wmma_kernel(const float* __restrict__ video,
                                 const _Float16* __restrict__ wh,
                                 const float* __restrict__ bias,
                                 float* __restrict__ tok_f32,
                                 _Float16* __restrict__ tok_h) {
  const int L  = threadIdx.x & 31;
  const int t0 = blockIdx.x * 16;
  const int e0 = (threadIdx.x >> 5) * 64;

  // decode the token this lane supplies for the B operand (n = L&15)
  const int t = t0 + (L & 15);
  const int b = t / NTOK;
  int rem = t % NTOK;
  const int d = rem / 196; rem %= 196;
  const int h = rem / 14;
  const int w = rem % 14;
  const long vbase = (long)b * 3 * 32 * 224 * 224;
  const int khalf16 = (L >> 4) * 16;

  v8f acc[4] = {};

  for (int k0 = 0; k0 < KCONV; k0 += 32) {
    // B fragment: K = k0 + khalf16 + i (i=0..15) -> fixed (c,kd,kh), kw=0..15
    const int koff = k0 + khalf16;      // multiple of 16
    const int q    = koff >> 4;         // (c*3+kd)*16 + kh
    const int kh   = q & 15;
    const int kdc  = q >> 4;
    const int kd   = kdc % 3;
    const int c    = kdc / 3;
    const int id   = 2 * d + kd - 1;    // depth index (pad=1, stride=2)
    v16h bfrag = {};
    if (id >= 0) {                      // id<32 always holds
      const float* p = video + vbase + ((long)c * 32 + id) * 224 * 224
                       + (h * 16 + kh) * 224 + w * 16;
      const float4* p4 = (const float4*)p;
#pragma unroll
      for (int j = 0; j < 4; ++j) {
        float4 f = p4[j];
        bfrag[4 * j + 0] = (_Float16)f.x;
        bfrag[4 * j + 1] = (_Float16)f.y;
        bfrag[4 * j + 2] = (_Float16)f.z;
        bfrag[4 * j + 3] = (_Float16)f.w;
      }
    }
    const int ka = k0 + (L >> 4) * 16;
#pragma unroll
    for (int tile = 0; tile < 4; ++tile) {
      const _Float16* ap = wh + (long)(e0 + tile * 16 + (L & 15)) * KCONV + ka;
      v8h alo = *(const v8h*)ap;
      v8h ahi = *(const v8h*)(ap + 8);
      v16h afrag;
#pragma unroll
      for (int j = 0; j < 8; ++j) { afrag[j] = alo[j]; afrag[j + 8] = ahi[j]; }
      acc[tile] = __builtin_amdgcn_wmma_f32_16x16x32_f16(
          false, afrag, false, bfrag, (short)0, acc[tile], false, false);
    }
  }
  // epilogue: D row m = channel, col n = token; bias + relu; write f32 & f16
  const int tw = t0 + (L & 15);
#pragma unroll
  for (int tile = 0; tile < 4; ++tile) {
#pragma unroll
    for (int r = 0; r < 8; ++r) {
      const int e = e0 + tile * 16 + r + 8 * (L >> 4);
      float v = acc[tile][r] + bias[e];
      v = fmaxf(v, 0.0f);
      tok_f32[(long)tw * EDIM + e] = v;
      tok_h[(long)tw * EDIM + e]   = (_Float16)v;
    }
  }
}

// ------------------------------------------- generic 256-K WMMA GEMM: A@W.T
// A (M,256) f16 row-major, W (256,256) f16 row-major (out-feature rows).
// flags: 1=f32 out, 2=f16 out, 4=+addvec residual, 8=mask rows by count[s]
__global__ void gemm256_wmma_kernel(const _Float16* __restrict__ A,
                                    const _Float16* __restrict__ W,
                                    const float* __restrict__ bias,
                                    float* __restrict__ outF32,
                                    _Float16* __restrict__ outF16,
                                    const float* __restrict__ addvec,
                                    const int* __restrict__ count,
                                    int flags) {
  const int L  = threadIdx.x & 31;
  const int m0 = blockIdx.x * 16;
  const int n0 = (threadIdx.x >> 5) * 64;

  v8f acc[4] = {};
  const _Float16* arow = A + (long)(m0 + (L & 15)) * 256;
  const int bn = L & 15;
  const int kb = L >> 4;

  for (int k0 = 0; k0 < 256; k0 += 32) {
    const int ka = k0 + kb * 16;
    v8h alo = *(const v8h*)(arow + ka);
    v8h ahi = *(const v8h*)(arow + ka + 8);
    v16h afrag;
#pragma unroll
    for (int j = 0; j < 8; ++j) { afrag[j] = alo[j]; afrag[j + 8] = ahi[j]; }
    __builtin_prefetch(arow + ka + 32, 0, 1);   // global_prefetch_b8
    const int kw = k0 + kb * 16;
#pragma unroll
    for (int tile = 0; tile < 4; ++tile) {
      const _Float16* wp = W + (long)(n0 + tile * 16 + bn) * 256 + kw;
      v16h bfrag = *(const v16h*)wp;
      acc[tile] = __builtin_amdgcn_wmma_f32_16x16x32_f16(
          false, afrag, false, bfrag, (short)0, acc[tile], false, false);
    }
  }
#pragma unroll
  for (int tile = 0; tile < 4; ++tile) {
#pragma unroll
    for (int r = 0; r < 8; ++r) {
      const int m = m0 + r + 8 * (L >> 4);
      const int n = n0 + tile * 16 + (L & 15);
      float v = acc[tile][r] + bias[n];
      if (flags & 4) v += addvec[n];
      if (flags & 8) { int s = m % NTOK; if (count[s] <= 0) v = 0.0f; }
      const long o = (long)m * 256 + n;
      if (flags & 1) outF32[o] = v;
      if (flags & 2) outF16[o] = (_Float16)v;
    }
  }
}

// --------------------------------------------------------- row LayerNorm
__global__ void ln_rows_kernel(const float* __restrict__ x,
                               const float* __restrict__ g,
                               const float* __restrict__ be,
                               _Float16* __restrict__ out) {
  const int L   = threadIdx.x & 31;
  const int row = blockIdx.x * (blockDim.x >> 5) + (threadIdx.x >> 5);
  const float* xr = x + (long)row * 256;
  float vals[8], s = 0.f, s2 = 0.f;
#pragma unroll
  for (int j = 0; j < 8; ++j) {
    float v = xr[j * 32 + L];
    vals[j] = v; s += v; s2 += v * v;
  }
#pragma unroll
  for (int o = 16; o > 0; o >>= 1) {
    s  += __shfl_xor(s, o, 32);
    s2 += __shfl_xor(s2, o, 32);
  }
  const float mean = s * (1.f / 256.f);
  const float var  = s2 * (1.f / 256.f) - mean * mean;
  const float rstd = rsqrtf(var + 1e-5f);
#pragma unroll
  for (int j = 0; j < 8; ++j) {
    const int e = j * 32 + L;
    out[(long)row * 256 + e] = (_Float16)((vals[j] - mean) * rstd * g[e] + be[e]);
  }
}

// ---------------------------------------- token entropy -> boundary bits
__global__ void entropy_kernel(const float* __restrict__ tok,
                               const float* __restrict__ table,
                               int* __restrict__ boundary) {
  const int t = blockIdx.x * blockDim.x + threadIdx.x;
  if (t >= NTOK) return;
  const float* xr = tok + (long)t * 256;   // batch-0 tokens are rows [0,3136)
  float s = 0.f;
  for (int e = 0; e < 256; ++e) s += xr[e];
  float rb = rintf(s * (1.f / 256.f) * 255.f);   // round-half-even
  rb = fminf(fmaxf(rb, 0.f), 255.f);
  const float* lg = table + (int)rb * 256;
  float m = -1e30f;
  for (int e = 0; e < 256; ++e) m = fmaxf(m, lg[e]);
  float den = 0.f;
  for (int e = 0; e < 256; ++e) den += __expf(lg[e] - m);
  const float inv = 1.f / den;
  float ent = 0.f;
  for (int e = 0; e < 256; ++e) {
    float p = __expf(lg[e] - m) * inv;
    ent -= p * __log2f(p + 1e-9f);
  }
  boundary[t] = (ent > 1.5f) ? 1 : 0;
}

// ------------------------------- sequential segment scan + start/end/count
__global__ void seg_scan_kernel(const int* __restrict__ boundary,
                                int* __restrict__ segStart,
                                int* __restrict__ segEnd,
                                int* __restrict__ count) {
  for (int i = threadIdx.x; i < NTOK; i += blockDim.x) {
    segStart[i] = 0; segEnd[i] = 0; count[i] = 0;
  }
  __syncthreads();
  if (threadIdx.x == 0) {
    int run = 0;
    for (int n = 0; n < NTOK; ++n) {
      const int s = run;                 // seg[n] = exclusive prefix sum
      if (count[s] == 0) segStart[s] = n;
      segEnd[s] = n + 1;
      count[s]++;
      run += boundary[n];
    }
  }
}

// -------------------------------------- tiny q path: LN(q0@Wq+b)@Wqi*scale
__global__ void qp_kernel(const float* __restrict__ gq,
                          const float* __restrict__ wq_w,
                          const float* __restrict__ wq_b,
                          const float* __restrict__ lnq_g,
                          const float* __restrict__ lnq_b,
                          const float* __restrict__ in_proj_w,
                          const float* __restrict__ in_proj_b,
                          float* __restrict__ qp) {
  __shared__ float shv[256];
  __shared__ float stats[2];
  const int e = threadIdx.x;
  shv[e] = gq[e];
  __syncthreads();
  float acc = wq_b[e];
  for (int k = 0; k < 256; ++k) acc += shv[k] * wq_w[e * 256 + k];
  __syncthreads();
  shv[e] = acc;
  __syncthreads();
  if (e == 0) {
    float s = 0.f, s2 = 0.f;
    for (int k = 0; k < 256; ++k) { s += shv[k]; s2 += shv[k] * shv[k]; }
    const float mean = s / 256.f;
    stats[0] = mean;
    stats[1] = rsqrtf(s2 / 256.f - mean * mean + 1e-5f);
  }
  __syncthreads();
  const float ln = (acc - stats[0]) * stats[1] * lnq_g[e] + lnq_b[e];
  __syncthreads();
  shv[e] = ln;
  __syncthreads();
  float acc2 = in_proj_b[e];                 // Wqi = in_proj_w rows [0,256)
  for (int k = 0; k < 256; ++k) acc2 += shv[k] * in_proj_w[e * 256 + k];
  qp[e] = acc2 * 0.125f;                     // dh^-0.5 = 1/8
}

// -------------------------------------------- scores[t,h] = qp[h] . kp[t,h]
__global__ void scores_kernel(const float* __restrict__ qp,
                              const float* __restrict__ kp,
                              float* __restrict__ scores) {
  const int idx = blockIdx.x * blockDim.x + threadIdx.x;  // t*4 + h
  if (idx >= MTOT * NHEAD) return;
  const int t = idx >> 2, h = idx & 3;
  const float* kr = kp + (long)t * 256 + h * DHEAD;
  const float* qr = qp + h * DHEAD;
  float s = 0.f;
  for (int d2 = 0; d2 < DHEAD; ++d2) s += qr[d2] * kr[d2];
  scores[idx] = s;
}

// ------------------- segment softmax + weighted vp sum, one wave per (s,b,h)
__global__ void segattn_kernel(const float* __restrict__ scores,
                               const float* __restrict__ vp,
                               const int* __restrict__ segStart,
                               const int* __restrict__ segEnd,
                               _Float16* __restrict__ attn) {
  const int u = blockIdx.x * (blockDim.x >> 5) + (threadIdx.x >> 5);
  const int L = threadIdx.x & 31;
  const int s = u >> 4, r = u & 15, b = r >> 2, h = r & 3;
  const long orow = ((long)b * NTOK + s) * 256 + h * DHEAD;
  const int n0 = segStart[s], n1 = segEnd[s];
  if (n1 <= n0) {                     // empty segment -> attn = 0
    attn[orow + L]      = (_Float16)0.f;
    attn[orow + 32 + L] = (_Float16)0.f;
    return;
  }
  float m = -1e30f;
  for (int n = n0 + L; n < n1; n += 32)
    m = fmaxf(m, scores[((long)b * NTOK + n) * 4 + h]);
#pragma unroll
  for (int o = 16; o > 0; o >>= 1) m = fmaxf(m, __shfl_xor(m, o, 32));
  float den = 0.f, a0 = 0.f, a1 = 0.f;
  for (int n = n0; n < n1; ++n) {
    const float w = __expf(scores[((long)b * NTOK + n) * 4 + h] - m);
    const float* vr = vp + ((long)b * NTOK + n) * 256 + h * DHEAD;
    a0 += w * vr[L];
    a1 += w * vr[32 + L];
    den += w;
  }
  const float inv = 1.f / den;
  attn[orow + L]      = (_Float16)(a0 * inv);
  attn[orow + 32 + L] = (_Float16)(a1 * inv);
}

// ---------------------------------------------------------------- launcher
extern "C" void kernel_launch(void* const* d_in, const int* in_sizes, int n_in,
                              void* d_out, int out_size, void* d_ws, size_t ws_size,
                              hipStream_t stream) {
  const float* video      = (const float*)d_in[0];
  const float* conv_w     = (const float*)d_in[1];
  const float* conv_b     = (const float*)d_in[2];
  const float* wq_w       = (const float*)d_in[3];
  const float* wq_b       = (const float*)d_in[4];
  const float* wk_w       = (const float*)d_in[5];
  const float* wk_b       = (const float*)d_in[6];
  const float* wv_w       = (const float*)d_in[7];
  const float* wv_b       = (const float*)d_in[8];
  const float* lnq_g      = (const float*)d_in[9];
  const float* lnq_b      = (const float*)d_in[10];
  const float* lnk_g      = (const float*)d_in[11];
  const float* lnk_b      = (const float*)d_in[12];
  const float* lnv_g      = (const float*)d_in[13];
  const float* lnv_b      = (const float*)d_in[14];
  const float* in_proj_w  = (const float*)d_in[15];
  const float* in_proj_b  = (const float*)d_in[16];
  const float* out_proj_w = (const float*)d_in[17];
  const float* out_proj_b = (const float*)d_in[18];
  const float* dense_w    = (const float*)d_in[19];
  const float* dense_b    = (const float*)d_in[20];
  const float* bproj_w    = (const float*)d_in[21];
  const float* bproj_b    = (const float*)d_in[22];
  const float* group_q    = (const float*)d_in[23];
  const float* ent_table  = (const float*)d_in[24];
  float* out = (float*)d_out;

  // ---- workspace carve-up (256B aligned)
  char* ws = (char*)d_ws;
  size_t off = 0;
  auto carve = [&](size_t bytes) {
    void* p = ws + off;
    off += (bytes + 255) & ~(size_t)255;
    return p;
  };
  float*    tok_f32 = (float*)   carve((size_t)MTOT * EDIM * 4);
  _Float16* tok_h   = (_Float16*)carve((size_t)MTOT * EDIM * 2);
  _Float16* convw_h = (_Float16*)carve((size_t)EDIM * KCONV * 2);
  _Float16* wk_h    = (_Float16*)carve((size_t)EDIM * EDIM * 2);
  _Float16* wv_h    = (_Float16*)carve((size_t)EDIM * EDIM * 2);
  _Float16* wki_h   = (_Float16*)carve((size_t)EDIM * EDIM * 2);
  _Float16* wvi_h   = (_Float16*)carve((size_t)EDIM * EDIM * 2);
  _Float16* wo_h    = (_Float16*)carve((size_t)EDIM * EDIM * 2);
  _Float16* wd_h    = (_Float16*)carve((size_t)EDIM * EDIM * 2);
  _Float16* wb_h    = (_Float16*)carve((size_t)EDIM * EDIM * 2);
  float*    tmp_f32 = (float*)   carve((size_t)MTOT * EDIM * 4);  // k_pre / v_pre
  _Float16* kln_h   = (_Float16*)carve((size_t)MTOT * EDIM * 2);  // later: out2
  _Float16* vln_h   = (_Float16*)carve((size_t)MTOT * EDIM * 2);  // later: attn
  float*    kp_f32  = (float*)   carve((size_t)MTOT * EDIM * 4);
  float*    vp_f32  = (float*)   carve((size_t)MTOT * EDIM * 4);
  float*    scores  = (float*)   carve((size_t)MTOT * NHEAD * 4);
  int*      boundary= (int*)     carve((size_t)NTOK * 4);
  int*      segStart= (int*)     carve((size_t)NTOK * 4);
  int*      segEnd  = (int*)     carve((size_t)NTOK * 4);
  int*      count   = (int*)     carve((size_t)NTOK * 4);
  float*    qp      = (float*)   carve((size_t)EDIM * 4);
  // buffer reuse after their producers/consumers are done:
  _Float16* attn_h = vln_h;   // vln consumed by vp GEMM before segattn writes
  _Float16* out1_h = tok_h;   // tok_h consumed by k/v GEMMs before out-chain
  _Float16* out2_h = kln_h;   // kln consumed by kp GEMM before dense GEMM

  // ---- weight conversions to f16
  cvt_f16_kernel<<<256, 256, 0, stream>>>(conv_w, convw_h, EDIM * KCONV);
  cvt_f16_kernel<<<64, 256, 0, stream>>>(wk_w, wk_h, EDIM * EDIM);
  cvt_f16_kernel<<<64, 256, 0, stream>>>(wv_w, wv_h, EDIM * EDIM);
  cvt_f16_kernel<<<64, 256, 0, stream>>>(in_proj_w + EDIM * EDIM,     wki_h, EDIM * EDIM);
  cvt_f16_kernel<<<64, 256, 0, stream>>>(in_proj_w + 2 * EDIM * EDIM, wvi_h, EDIM * EDIM);
  cvt_f16_kernel<<<64, 256, 0, stream>>>(out_proj_w, wo_h, EDIM * EDIM);
  cvt_f16_kernel<<<64, 256, 0, stream>>>(dense_w,    wd_h, EDIM * EDIM);
  cvt_f16_kernel<<<64, 256, 0, stream>>>(bproj_w,    wb_h, EDIM * EDIM);

  // ---- conv + bias + relu -> tokens (f32 + f16), WMMA im2col
  conv_wmma_kernel<<<MTOT / 16, 128, 0, stream>>>(video, convw_h, conv_b,
                                                  tok_f32, tok_h);

  // ---- entropy boundaries and segment bookkeeping
  entropy_kernel<<<(NTOK + 255) / 256, 256, 0, stream>>>(tok_f32, ent_table, boundary);
  seg_scan_kernel<<<1, 256, 0, stream>>>(boundary, segStart, segEnd, count);

  // ---- q path (tiny)
  qp_kernel<<<1, 256, 0, stream>>>(group_q, wq_w, wq_b, lnq_g, lnq_b,
                                   in_proj_w, in_proj_b, qp);

  // ---- k path: GEMM -> LN -> GEMM
  gemm256_wmma_kernel<<<MTOT / 16, 128, 0, stream>>>(tok_h, wk_h, wk_b,
      tmp_f32, nullptr, nullptr, nullptr, 1);
  ln_rows_kernel<<<MTOT / 4, 128, 0, stream>>>(tmp_f32, lnk_g, lnk_b, kln_h);
  gemm256_wmma_kernel<<<MTOT / 16, 128, 0, stream>>>(kln_h, wki_h,
      in_proj_b + EDIM, kp_f32, nullptr, nullptr, nullptr, 1);

  // ---- v path: GEMM -> LN -> GEMM
  gemm256_wmma_kernel<<<MTOT / 16, 128, 0, stream>>>(tok_h, wv_h, wv_b,
      tmp_f32, nullptr, nullptr, nullptr, 1);
  ln_rows_kernel<<<MTOT / 4, 128, 0, stream>>>(tmp_f32, lnv_g, lnv_b, vln_h);
  gemm256_wmma_kernel<<<MTOT / 16, 128, 0, stream>>>(vln_h, wvi_h,
      in_proj_b + 2 * EDIM, vp_f32, nullptr, nullptr, nullptr, 1);

  // ---- attention scores + segmented softmax-weighted sum
  scores_kernel<<<(MTOT * NHEAD + 255) / 256, 256, 0, stream>>>(qp, kp_f32, scores);
  segattn_kernel<<<(NTOK * 16) / 4, 128, 0, stream>>>(scores, vp_f32,
                                                      segStart, segEnd, attn_h);

  // ---- output chain: out_proj -> dense(+q0 residual) -> bproj(masked)
  gemm256_wmma_kernel<<<MTOT / 16, 128, 0, stream>>>(attn_h, wo_h, out_proj_b,
      nullptr, out1_h, nullptr, nullptr, 2);
  gemm256_wmma_kernel<<<MTOT / 16, 128, 0, stream>>>(out1_h, wd_h, dense_b,
      nullptr, out2_h, group_q, nullptr, 2 | 4);
  gemm256_wmma_kernel<<<MTOT / 16, 128, 0, stream>>>(out2_h, wb_h, bproj_b,
      out, nullptr, nullptr, count, 1 | 8);
}
